// HGC_LRN_25237227832003
// MI455X (gfx1250) — matrix operand order repaired and checked by reading, not compiled, not run on hardware
//
#include <hip/hip_runtime.h>
#include <hip/hip_bf16.h>

// ---------------------------------------------------------------------------
// N=20000 nodes, S=4096 cols, E=640000 edges, D=256 features.
//   phi[S,D] = relu(idx @ w1 + b1) @ w2 + b2                  (tiny MLP)
//   h[N,D]   = (mask @ phi) / max(cnt,1)   (wave32 ballot-compacted scan,
//                                           bandwidth-bound on 328MB init)
//   3x: h_agg[col] += h[row]*attr          (float4 gather + f32 atomics in L2)
//       h = relu( bf16(h_agg) @ W )        (v_wmma_f32_16x16x32_bf16)
// ---------------------------------------------------------------------------

typedef __attribute__((ext_vector_type(16))) __bf16 v16bf;
typedef __attribute__((ext_vector_type(8)))  float  v8f;

#define DFEAT 256
#define SCOLS 4096

__device__ __forceinline__ unsigned short f2bf_rne(float x) {
    unsigned int u = __float_as_uint(x);
    unsigned int r = u + 0x7FFFu + ((u >> 16) & 1u);   // round-to-nearest-even
    return (unsigned short)(r >> 16);
}

// ---------------- phi[j,d] = relu(j*w1+b1) @ w2 + b2 -----------------------
__global__ void phi_kernel(const float* __restrict__ w1, const float* __restrict__ b1,
                           const float* __restrict__ w2, const float* __restrict__ b2,
                           float* __restrict__ phi) {
    const int j = blockIdx.x;       // 0..S-1
    const int d = threadIdx.x;      // 0..255
    const float x = (float)j;
    float acc = b2[d];
#pragma unroll
    for (int k = 0; k < 16; ++k) {
        float hk = fmaxf(fmaf(x, w1[k], b1[k]), 0.0f);
        acc = fmaf(hk, w2[k * DFEAT + d], acc);
    }
    phi[(size_t)j * DFEAT + d] = acc;
}

// ---------------- Wt[n,k] = bf16(W[k,n])  (B operand, n-major) -------------
__global__ void make_wt_kernel(const float* __restrict__ W, unsigned short* __restrict__ Wt) {
    const int k = blockIdx.x;   // 0..255
    const int n = threadIdx.x;  // 0..255
    Wt[(size_t)n * DFEAT + k] = f2bf_rne(W[(size_t)k * DFEAT + n]);
}

// ---------------- masked mean: wave32 ballot compaction --------------------
// One 256-thread block (8 waves) per row. Wave w scans columns
// [w*512, w*512+512) with coalesced loads; ballot+rank writes the nonzero
// column indices into LDS in ascending column order (deterministic, no
// atomics). Then all 256 threads (one per feature d) accumulate phi rows
// for the ~2% compacted indices. Instruction cost ~2.5k wave-inst per row,
// keeping the kernel near the 23.3 TB/s roofline for the 328MB init read.
__global__ void __launch_bounds__(256)
masked_mean_kernel(const float* __restrict__ init,
                   const float* __restrict__ phi,
                   float* __restrict__ h) {
    __shared__ int sidx[SCOLS];     // worst case: all columns nonzero
    __shared__ int swcnt[8];
    const int tid  = threadIdx.x;
    const int lane = tid & 31;
    const int wave = tid >> 5;
    const float* row = init + (size_t)blockIdx.x * SCOLS;

    const unsigned lt_mask = (1u << lane) - 1u;
    int cw = 0;                                  // per-wave compacted count
#pragma unroll 1
    for (int it = 0; it < 512 / 32; ++it) {
        const int j = wave * 512 + it * 32 + lane;     // coalesced
        const float v = row[j];
        const bool nz = (v != 0.0f);
        const unsigned m = __builtin_amdgcn_ballot_w32(nz);
        if (nz) {
            const int r = __popc(m & lt_mask);         // rank within wave
            sidx[wave * 512 + cw + r] = j;
        }
        cw += __popc(m);
    }
    if (lane == 0) swcnt[wave] = cw;
    __syncthreads();

    float acc = 0.0f;
    int   cnt = 0;
#pragma unroll 1
    for (int w = 0; w < 8; ++w) {
        const int c = swcnt[w];
        cnt += c;
        const int* sp = &sidx[w * 512];
#pragma unroll 1
        for (int t = 0; t < c; ++t) {                  // LDS broadcast index
            acc += phi[(size_t)sp[t] * DFEAT + tid];   // coalesced phi row
        }
    }
    const float inv = 1.0f / fmaxf((float)cnt, 1.0f);
    h[(size_t)blockIdx.x * DFEAT + tid] = acc * inv;
}

// ---------------- edge scatter: h_agg[col] += h[row] * attr ----------------
// 4 edges per 256-thread block; 64 lanes x float4 per edge -> b128 gathers.
__global__ void __launch_bounds__(256)
scatter_kernel(const float4* __restrict__ h4,
               const int* __restrict__ ei,
               const float* __restrict__ attr,
               float* __restrict__ hagg, int E) {
    const int tid = threadIdx.x;
    const int e   = blockIdx.x * 4 + (tid >> 6);
    const int t   = tid & 63;                    // float4 slot: d = 4t..4t+3
    const int row = ei[e];
    const int col = ei[E + e];
    const float a = attr[e];
    const float4 v = h4[(size_t)row * (DFEAT / 4) + t];
    float* dst = hagg + (size_t)col * DFEAT + t * 4;
    atomicAdd(dst + 0, v.x * a);
    atomicAdd(dst + 1, v.y * a);
    atomicAdd(dst + 2, v.z * a);
    atomicAdd(dst + 3, v.w * a);
}

// ---------------- fp32 -> bf16 convert ------------------------------------
__global__ void f32_to_bf16_kernel(const float* __restrict__ x,
                                   unsigned short* __restrict__ y, int n) {
    int i = blockIdx.x * blockDim.x + threadIdx.x;
    if (i < n) y[i] = f2bf_rne(x[i]);
}

// ---------------- WMMA GEMM:  C = relu(A[M,256] @ W[256,256]) --------------
// A: bf16 row-major [M,256]; Bt: bf16 [256,256] with Bt[n*256+k] = W[k][n].
// Block: 128 threads = 4 waves. Block tile: 32 M x 256 N.
// Wave w: cols [w*64, w*64+64), 2 m-subtiles x 4 n-subtiles -> 8 accumulators.
__global__ void __launch_bounds__(128)
gemm_relu_kernel(const unsigned short* __restrict__ A,
                 const unsigned short* __restrict__ Bt,
                 float* __restrict__ C, int M) {
    const int lane = threadIdx.x & 31;
    const int wave = threadIdx.x >> 5;          // 0..3
    const int m0   = blockIdx.x * 32;
    const int n0   = wave * 64;
    const int half = lane >> 4;                 // 0/1
    const int l16  = lane & 15;

    const unsigned int* A32 = (const unsigned int*)A;   // 2 bf16 per dword
    const unsigned int* B32 = (const unsigned int*)Bt;

    v8f zero = {};
    v8f acc[2][4];
#pragma unroll
    for (int i = 0; i < 2; ++i)
#pragma unroll
        for (int j = 0; j < 4; ++j) acc[i][j] = zero;

#pragma unroll 1
    for (int kb = 0; kb < DFEAT; kb += 32) {
        // A fragments (16x32 bf16): lane -> M, VGPR v -> K pair
        union { unsigned int u[8]; v16bf v; } afrag[2];
#pragma unroll
        for (int mt = 0; mt < 2; ++mt) {
            const int m = m0 + mt * 16 + l16;
#pragma unroll
            for (int r = 0; r < 8; ++r) {
                const int k = kb + ((r >> 2) << 4) + (half << 3) + ((r & 3) << 1);
                afrag[mt].u[r] = A32[m * (DFEAT / 2) + (k >> 1)];
            }
        }
#pragma unroll
        for (int nt = 0; nt < 4; ++nt) {
            // B fragment (32x16 bf16): lane%16 -> N, half/VGPR -> K pair
            union { unsigned int u[8]; v16bf v; } bfrag;
            const int n = n0 + nt * 16 + l16;
#pragma unroll
            for (int r = 0; r < 8; ++r) {
                const int k = kb + ((r >> 2) << 4) + (half << 3) + ((r & 3) << 1);
                bfrag.u[r] = B32[n * (DFEAT / 2) + (k >> 1)];
            }
#pragma unroll
            for (int mt = 0; mt < 2; ++mt) {
                acc[mt][nt] = __builtin_amdgcn_wmma_f32_16x16x32_bf16(
                    false, afrag[mt].v, false, bfrag.v,
                    (short)0, acc[mt][nt], false, false);
            }
        }
    }

    // Epilogue: ReLU + store.  C/D layout: VGPR r -> M = r + 8*half, N = lane%16
#pragma unroll
    for (int mt = 0; mt < 2; ++mt) {
#pragma unroll
        for (int nt = 0; nt < 4; ++nt) {
#pragma unroll
            for (int r = 0; r < 8; ++r) {
                const int m = m0 + mt * 16 + r + half * 8;
                const int n = n0 + nt * 16 + l16;
                C[(size_t)m * DFEAT + n] = fmaxf(acc[mt][nt][r], 0.0f);
            }
        }
    }
}

// ---------------------------------------------------------------------------
extern "C" void kernel_launch(void* const* d_in, const int* in_sizes, int n_in,
                              void* d_out, int out_size, void* d_ws, size_t ws_size,
                              hipStream_t stream) {
    const float* init = (const float*)d_in[0];
    const int*   ei   = (const int*)  d_in[1];
    const float* attr = (const float*)d_in[2];
    const float* w1   = (const float*)d_in[3];
    const float* b1   = (const float*)d_in[4];
    const float* w2   = (const float*)d_in[5];
    const float* b2   = (const float*)d_in[6];
    const float* W    = (const float*)d_in[7];
    float* out = (float*)d_out;

    const int E = in_sizes[2];                 // 640000
    const int N = in_sizes[0] / SCOLS;         // 20000
    const int D = DFEAT;

    // workspace carve-up (~55 MB)
    char* ws = (char*)d_ws;
    float*          phi  = (float*)ws;          ws += (size_t)SCOLS * D * sizeof(float);
    float*          h    = (float*)ws;          ws += (size_t)N * D * sizeof(float);
    float*          hagg = (float*)ws;          ws += (size_t)N * D * sizeof(float);
    unsigned short* Abf  = (unsigned short*)ws; ws += (size_t)N * D * sizeof(unsigned short);
    unsigned short* Wt   = (unsigned short*)ws; ws += (size_t)D * D * sizeof(unsigned short);

    phi_kernel<<<SCOLS, D, 0, stream>>>(w1, b1, w2, b2, phi);
    make_wt_kernel<<<D, D, 0, stream>>>(W, Wt);
    masked_mean_kernel<<<N, 256, 0, stream>>>(init, phi, h);

    const int nElem = N * D;
    for (int layer = 0; layer < 3; ++layer) {
        hipMemsetAsync(hagg, 0, (size_t)N * D * sizeof(float), stream);
        scatter_kernel<<<E / 4, 256, 0, stream>>>((const float4*)h, ei, attr, hagg, E);
        f32_to_bf16_kernel<<<(nElem + 255) / 256, 256, 0, stream>>>(hagg, Abf, nElem);
        float* dst = (layer == 2) ? out : h;
        gemm_relu_kernel<<<N / 32, 128, 0, stream>>>(Abf, Wt, dst, N);
    }
}
